// CrossWinAttention_63642825392267
// MI455X (gfx1250) — compile-verified, matrix operand order checked by reading, not run on hardware
//
#include <hip/hip_runtime.h>
#include <cstdint>
#include <cstddef>

// ---- problem constants ----
#define T_TOKENS 9216   // L(16) * Qn(576)
#define QN       576    // n*W1*W2
#define LWIN     16     // X*Y
#define DIMC     256
#define INNERC   1024
#define NHEADS   16
#define DHEAD    64

typedef __attribute__((ext_vector_type(16))) __bf16 v16bf;
typedef __attribute__((ext_vector_type(8)))  __bf16 v8bf;
typedef __attribute__((ext_vector_type(8)))  float  v8f;

// ---------------------------------------------------------------------------
// WMMA helpers (CDNA5 wave32, 16x16x32 bf16 -> f32)
// A-operand layout (ISA 7.12.2): lane m (m=lane&15) holds row m; per lane two
// contiguous 8-element (16B) chunks at K = half*8 and K = 16+half*8.
// B-operand is the dual with lane = output column (so we store B transposed,
// row = N, contiguous over K).
// ---------------------------------------------------------------------------
__device__ __forceinline__ v16bf load_frag(const __bf16* rowbase, int stride,
                                           int row, int k0, int hl) {
  const __bf16* p = rowbase + (size_t)row * stride + k0 + hl * 8;
  v8bf c0 = *(const v8bf*)p;
  v8bf c1 = *(const v8bf*)(p + 16);
  return __builtin_shufflevector(c0, c1, 0,1,2,3,4,5,6,7,8,9,10,11,12,13,14,15);
}

__device__ __forceinline__ v8f wmma_bf16(v16bf a, v16bf b, v8f c) {
  return __builtin_amdgcn_wmma_f32_16x16x32_bf16(false, a, false, b,
                                                 (short)0, c, false, false);
}

__device__ __forceinline__ v8f vzero8() {
  v8f z = {0.f,0.f,0.f,0.f,0.f,0.f,0.f,0.f};
  return z;
}

__device__ __forceinline__ float half_max(float v) {
  #pragma unroll
  for (int m = 1; m < 16; m <<= 1) v = fmaxf(v, __shfl_xor(v, m, 32));
  return v;
}
__device__ __forceinline__ float half_sum(float v) {
  #pragma unroll
  for (int m = 1; m < 16; m <<= 1) v += __shfl_xor(v, m, 32);
  return v;
}
__device__ __forceinline__ float wave_sum32(float v) {
  #pragma unroll
  for (int m = 1; m < 32; m <<= 1) v += __shfl_xor(v, m, 32);
  return v;
}

// ---------------------------------------------------------------------------
// K0: convert weights to bf16, transposed to N-major (B-operand friendly).
// grid (1024, 4) x 256
// ---------------------------------------------------------------------------
__global__ __launch_bounds__(256) void wcvt_kernel(
    const float* __restrict__ Wq, const float* __restrict__ Wk,
    const float* __restrict__ Wv, const float* __restrict__ Wp,
    __bf16* __restrict__ WqT, __bf16* __restrict__ WkT,
    __bf16* __restrict__ WvT, __bf16* __restrict__ WpT) {
  int sel = blockIdx.y;
  int idx = blockIdx.x * 256 + threadIdx.x;      // 0 .. 262143
  if (sel < 3) {
    const float* W = (sel == 0) ? Wq : (sel == 1) ? Wk : Wv;   // [256][1024]
    __bf16* D      = (sel == 0) ? WqT : (sel == 1) ? WkT : WvT; // [1024][256]
    int n = idx >> 8, k = idx & 255;
    D[idx] = (__bf16)W[k * INNERC + n];
  } else {
    int n = idx >> 10, c = idx & 1023;           // WpT [256][1024], Wp [1024][256]
    WpT[idx] = (__bf16)Wp[c * DIMC + n];
  }
}

// ---------------------------------------------------------------------------
// K1: LayerNorm + to_tokens gather -> bf16 token matrices [T][256].
// One wave per token. grid (1152, 3) x 256
// ---------------------------------------------------------------------------
__global__ __launch_bounds__(256) void ln_kernel(
    const float* __restrict__ q, const float* __restrict__ k,
    const float* __restrict__ v,
    const float* __restrict__ gq, const float* __restrict__ bq,
    const float* __restrict__ gk, const float* __restrict__ bk,
    const float* __restrict__ gv, const float* __restrict__ bv,
    __bf16* __restrict__ Xq, __bf16* __restrict__ Xk, __bf16* __restrict__ Xv) {
  int sel = blockIdx.y;
  const float* src = (sel == 0) ? q : (sel == 1) ? k : v;
  const float* g   = (sel == 0) ? gq : (sel == 1) ? gk : gv;
  const float* be  = (sel == 0) ? bq : (sel == 1) ? bk : bv;
  __bf16* dst      = (sel == 0) ? Xq : (sel == 1) ? Xk : Xv;

  int wid = threadIdx.x >> 5, lane = threadIdx.x & 31;
  int t = blockIdx.x * 8 + wid;                 // token index (l, qpos)
  int l = t / QN, qpos = t - l * QN;
  int x = l >> 2, y = l & 3;
  int n = qpos / 144, r = qpos - n * 144;
  int w1 = r / 12, w2 = r - w1 * 12;
  const float* row = src + (size_t)(((((n * 4 + x) * 4 + y) * 12 + w1) * 12 + w2)) * DIMC;

  const float4* p4 = (const float4*)(row + lane * 8);
  float4 u0 = p4[0], u1 = p4[1];
  float xv8[8] = {u0.x,u0.y,u0.z,u0.w,u1.x,u1.y,u1.z,u1.w};

  float s = 0.f;
  #pragma unroll
  for (int i = 0; i < 8; i++) s += xv8[i];
  s = wave_sum32(s);
  float mu = s * (1.f / 256.f);
  float vs = 0.f;
  #pragma unroll
  for (int i = 0; i < 8; i++) { float d = xv8[i] - mu; vs += d * d; }
  vs = wave_sum32(vs);
  float rstd = rsqrtf(vs * (1.f / 256.f) + 1e-5f);

  const float4* g4 = (const float4*)(g + lane * 8);
  const float4* b4 = (const float4*)(be + lane * 8);
  float4 g0 = g4[0], g1 = g4[1], e0 = b4[0], e1 = b4[1];
  float gg[8] = {g0.x,g0.y,g0.z,g0.w,g1.x,g1.y,g1.z,g1.w};
  float ee[8] = {e0.x,e0.y,e0.z,e0.w,e1.x,e1.y,e1.z,e1.w};

  v8bf o;
  #pragma unroll
  for (int i = 0; i < 8; i++)
    o[i] = (__bf16)((xv8[i] - mu) * rstd * gg[i] + ee[i]);
  *(v8bf*)(dst + (size_t)t * DIMC + lane * 8) = o;
}

// ---------------------------------------------------------------------------
// K2: QKV projection GEMM (WMMA) + bias + optional RoPE.
// Wave = 16 tokens x 64 cols (one head). mode 0: rope, row-major store (q/k).
// mode 1: no rope, d-major (transposed) store (v). grid (72, 16) x 256
// ---------------------------------------------------------------------------
__global__ __launch_bounds__(256) void qkv_gemm_kernel(
    const __bf16* __restrict__ X, const __bf16* __restrict__ WT,
    const float* __restrict__ bias, const float* __restrict__ rope,
    __bf16* __restrict__ dstRow, __bf16* __restrict__ dstT, int mode) {
  int wid = threadIdx.x >> 5, lane = threadIdx.x & 31;
  int ln_ = lane & 15, hl = lane >> 4;
  int mtile = blockIdx.x * 8 + wid;
  int tbase = mtile * 16;
  int head = blockIdx.y;

  v8f acc[4];
  #pragma unroll
  for (int j = 0; j < 4; j++) acc[j] = vzero8();

  const __bf16* Xr = X + (size_t)tbase * DIMC;
  #pragma unroll
  for (int kk = 0; kk < DIMC; kk += 32) {
    v16bf a = load_frag(Xr, DIMC, ln_, kk, hl);
    #pragma unroll
    for (int j = 0; j < 4; j++) {
      v16bf b = load_frag(WT + (size_t)(head * DHEAD + j * 16) * DIMC, DIMC, ln_, kk, hl);
      acc[j] = wmma_bf16(a, b, acc[j]);
    }
  }

  // bias
  #pragma unroll
  for (int j = 0; j < 4; j++) {
    float bv = bias[head * DHEAD + j * 16 + ln_];
    #pragma unroll
    for (int e = 0; e < 8; e++) acc[j][e] += bv;
  }

  int l = tbase / QN;
  int qbase = tbase - l * QN;

  if (mode == 0) {
    // RoPE: column d pairs with d+-32 => accumulator j pairs with j+-2, same lane.
    float res[4][8];
    #pragma unroll
    for (int j = 0; j < 4; j++) {
      int dp = j * 16 + ln_;
      #pragma unroll
      for (int e = 0; e < 8; e++) {
        int M = e + hl * 8;
        int qpos = qbase + M;
        float f = rope[(size_t)qpos * DHEAD + dp];
        float cf = cosf(f), sf = sinf(f);
        float partner = (j < 2) ? -acc[j + 2][e] : acc[j - 2][e];
        res[j][e] = acc[j][e] * cf + partner * sf;
      }
    }
    #pragma unroll
    for (int j = 0; j < 4; j++) {
      int dp = j * 16 + ln_;
      #pragma unroll
      for (int e = 0; e < 8; e++) {
        int M = e + hl * 8;
        dstRow[((size_t)head * T_TOKENS + tbase + M) * DHEAD + dp] = (__bf16)res[j][e];
      }
    }
  } else {
    // V: store transposed [head][l][d][key] for PV B-operand.
    #pragma unroll
    for (int j = 0; j < 4; j++) {
      int dp = j * 16 + ln_;
      #pragma unroll
      for (int e = 0; e < 8; e++) {
        int M = e + hl * 8;
        int qpos = qbase + M;
        dstT[(((size_t)head * LWIN + l) * DHEAD + dp) * QN + qpos] = (__bf16)acc[j][e];
      }
    }
  }
}

// ---------------------------------------------------------------------------
// K3: flash attention, one wave per (qtile, window, head).
// grid (36, 16, 16) x 32
// ---------------------------------------------------------------------------
__global__ __launch_bounds__(32) void attn_kernel(
    const __bf16* __restrict__ Qb, const __bf16* __restrict__ Kb,
    const __bf16* __restrict__ Vt, const float* __restrict__ head_gate,
    const float* __restrict__ als, __bf16* __restrict__ Ab) {
  __shared__ __bf16 ldsP[16 * 40];   // padded stride to spread LDS banks

  int lane = threadIdx.x, ln_ = lane & 15, hl = lane >> 4;
  int qbase = blockIdx.x * 16;
  int l = blockIdx.y, h = blockIdx.z;

  const __bf16* Qr = Qb + ((size_t)h * T_TOKENS + (size_t)l * QN + qbase) * DHEAD;
  const __bf16* Kr = Kb + ((size_t)h * T_TOKENS + (size_t)l * QN) * DHEAD;
  const __bf16* Vr = Vt + ((size_t)(h * LWIN + l) * DHEAD) * QN;

  float gate = fminf(fmaxf(head_gate[h], 0.f), 1.f);
  float sfac = gate * (als[h] - 2.0794415416798357f);  // log(64^-0.5)

  v16bf aq0 = load_frag(Qr, DHEAD, ln_, 0, hl);
  v16bf aq1 = load_frag(Qr, DHEAD, ln_, 32, hl);

  v8f o[4];
  #pragma unroll
  for (int j = 0; j < 4; j++) o[j] = vzero8();
  float mi[8], li[8];
  #pragma unroll
  for (int e = 0; e < 8; e++) { mi[e] = -3.0e38f; li[e] = 0.f; }

  for (int kb = 0; kb < QN; kb += 32) {
    v8f s0 = vzero8(), s1 = vzero8();
    {
      v16bf b00 = load_frag(Kr + (size_t)kb * DHEAD, DHEAD, ln_, 0, hl);
      v16bf b01 = load_frag(Kr + (size_t)kb * DHEAD, DHEAD, ln_, 32, hl);
      s0 = wmma_bf16(aq0, b00, s0);
      s0 = wmma_bf16(aq1, b01, s0);
      v16bf b10 = load_frag(Kr + (size_t)(kb + 16) * DHEAD, DHEAD, ln_, 0, hl);
      v16bf b11 = load_frag(Kr + (size_t)(kb + 16) * DHEAD, DHEAD, ln_, 32, hl);
      s1 = wmma_bf16(aq0, b10, s1);
      s1 = wmma_bf16(aq1, b11, s1);
    }
    float p0[8], p1[8];
    #pragma unroll
    for (int e = 0; e < 8; e++) {
      float v0 = s0[e] * sfac, v1 = s1[e] * sfac;
      float rm = half_max(fmaxf(v0, v1));
      float mn = fmaxf(mi[e], rm);
      float rsc = __expf(mi[e] - mn);
      mi[e] = mn;
      li[e] *= rsc;
      o[0][e] *= rsc; o[1][e] *= rsc; o[2][e] *= rsc; o[3][e] *= rsc;
      float e0 = __expf(v0 - mn), e1 = __expf(v1 - mn);
      p0[e] = e0; p1[e] = e1;
      li[e] += half_sum(e0 + e1);
    }
    // reshape P (C-layout) -> A-operand layout via LDS
    #pragma unroll
    for (int e = 0; e < 8; e++) {
      int M = e + hl * 8;
      ldsP[M * 40 + ln_]      = (__bf16)p0[e];
      ldsP[M * 40 + 16 + ln_] = (__bf16)p1[e];
    }
    __syncthreads();
    v16bf ap = load_frag(ldsP, 40, ln_, 0, hl);
    #pragma unroll
    for (int j = 0; j < 4; j++) {
      v16bf bv = load_frag(Vr + (size_t)(j * 16) * QN, QN, ln_, kb, hl);
      o[j] = wmma_bf16(ap, bv, o[j]);
    }
    __syncthreads();
  }

  #pragma unroll
  for (int e = 0; e < 8; e++) {
    float inv = 1.f / li[e];
    int M = e + hl * 8;
    size_t base = ((size_t)h * T_TOKENS + (size_t)l * QN + qbase + M) * DHEAD;
    #pragma unroll
    for (int j = 0; j < 4; j++)
      Ab[base + j * 16 + ln_] = (__bf16)(o[j][e] * inv);
  }
}

// ---------------------------------------------------------------------------
// K4: output projection GEMM (WMMA): Z[t][256] = A[t][1024] @ Wp + bp.
// grid (72, 16) x 256
// ---------------------------------------------------------------------------
__global__ __launch_bounds__(256) void out_gemm_kernel(
    const __bf16* __restrict__ Ab, const __bf16* __restrict__ WpT,
    const float* __restrict__ bp, float* __restrict__ Z) {
  int wid = threadIdx.x >> 5, lane = threadIdx.x & 31;
  int ln_ = lane & 15, hl = lane >> 4;
  int mtile = blockIdx.x * 8 + wid;
  int tbase = mtile * 16;
  int nt = blockIdx.y;

  v8f acc = vzero8();
  #pragma unroll 4
  for (int c = 0; c < INNERC; c += 32) {
    // feature c = h*64 + d  -> gather from per-head attention output
    const __bf16* Ar = Ab + ((size_t)(c >> 6) * T_TOKENS + tbase) * DHEAD + (c & 63);
    v16bf a = load_frag(Ar, DHEAD, ln_, 0, hl);
    v16bf b = load_frag(WpT + (size_t)(nt * 16) * INNERC, INNERC, ln_, c, hl);
    acc = wmma_bf16(a, b, acc);
  }
  float bias = bp[nt * 16 + ln_];
  #pragma unroll
  for (int e = 0; e < 8; e++) {
    int M = e + hl * 8;
    Z[(size_t)(tbase + M) * DIMC + nt * 16 + ln_] = acc[e] + bias;
  }
}

// ---------------------------------------------------------------------------
// K5: mean over n (4 values) + skip. grid 2304 x 256
// ---------------------------------------------------------------------------
__global__ __launch_bounds__(256) void finalize_kernel(
    const float* __restrict__ Z, const float* __restrict__ skip,
    float* __restrict__ out) {
  int gid = blockIdx.x * 256 + threadIdx.x;
  if (gid >= 4 * 4 * 12 * 12 * DIMC) return;
  int d = gid & 255;
  int rem = gid >> 8;
  int w2 = rem % 12; rem /= 12;
  int w1 = rem % 12; rem /= 12;
  int y = rem & 3, x = rem >> 2;
  int l = x * 4 + y;
  int qp0 = w1 * 12 + w2;
  float s = 0.f;
  #pragma unroll
  for (int n = 0; n < 4; n++)
    s += Z[((size_t)l * QN + n * 144 + qp0) * DIMC + d];
  out[gid] = 0.25f * s + skip[gid];
}

// ---------------------------------------------------------------------------
extern "C" void kernel_launch(void* const* d_in, const int* in_sizes, int n_in,
                              void* d_out, int out_size, void* d_ws, size_t ws_size,
                              hipStream_t stream) {
  const float* q    = (const float*)d_in[0];
  const float* k    = (const float*)d_in[1];
  const float* v    = (const float*)d_in[2];
  const float* skip = (const float*)d_in[3];
  const float* rope = (const float*)d_in[4];
  const float* hg   = (const float*)d_in[5];
  const float* lnqg = (const float*)d_in[6];
  const float* lnqb = (const float*)d_in[7];
  const float* lnkg = (const float*)d_in[8];
  const float* lnkb = (const float*)d_in[9];
  const float* lnvg = (const float*)d_in[10];
  const float* lnvb = (const float*)d_in[11];
  const float* Wq   = (const float*)d_in[12];
  const float* bq   = (const float*)d_in[13];
  const float* Wk   = (const float*)d_in[14];
  const float* bk   = (const float*)d_in[15];
  const float* Wv   = (const float*)d_in[16];
  const float* bv   = (const float*)d_in[17];
  const float* Wp   = (const float*)d_in[18];
  const float* bp   = (const float*)d_in[19];
  const float* als  = (const float*)d_in[20];

  // workspace carve-up (all sizes multiples of 256B)
  char* w = (char*)d_ws;
  auto alloc = [&](size_t bytes) {
    void* p = (void*)w;
    w += (bytes + 255) & ~(size_t)255;
    return p;
  };
  __bf16* WqT = (__bf16*)alloc((size_t)INNERC * DIMC * 2);
  __bf16* WkT = (__bf16*)alloc((size_t)INNERC * DIMC * 2);
  __bf16* WvT = (__bf16*)alloc((size_t)INNERC * DIMC * 2);
  __bf16* WpT = (__bf16*)alloc((size_t)DIMC * INNERC * 2);
  __bf16* Xq  = (__bf16*)alloc((size_t)T_TOKENS * DIMC * 2);
  __bf16* Xk  = (__bf16*)alloc((size_t)T_TOKENS * DIMC * 2);
  __bf16* Xv  = (__bf16*)alloc((size_t)T_TOKENS * DIMC * 2);
  __bf16* Qb  = (__bf16*)alloc((size_t)NHEADS * T_TOKENS * DHEAD * 2);
  __bf16* Kb  = (__bf16*)alloc((size_t)NHEADS * T_TOKENS * DHEAD * 2);
  __bf16* Vt  = (__bf16*)alloc((size_t)NHEADS * T_TOKENS * DHEAD * 2);
  __bf16* Ab  = (__bf16*)alloc((size_t)NHEADS * T_TOKENS * DHEAD * 2);
  // Z (f32, 9.4MB) reuses the Xq+Xk region (dead after qkv_gemm).
  float* Z = (float*)Xq;

  wcvt_kernel<<<dim3(1024, 4), 256, 0, stream>>>(Wq, Wk, Wv, Wp, WqT, WkT, WvT, WpT);
  ln_kernel<<<dim3(T_TOKENS / 8, 3), 256, 0, stream>>>(
      q, k, v, lnqg, lnqb, lnkg, lnkb, lnvg, lnvb, Xq, Xk, Xv);
  qkv_gemm_kernel<<<dim3(72, 16), 256, 0, stream>>>(Xq, WqT, bq, rope, Qb, nullptr, 0);
  qkv_gemm_kernel<<<dim3(72, 16), 256, 0, stream>>>(Xk, WkT, bk, rope, Kb, nullptr, 0);
  qkv_gemm_kernel<<<dim3(72, 16), 256, 0, stream>>>(Xv, WvT, bv, rope, nullptr, Vt, 1);
  attn_kernel<<<dim3(36, 16, 16), 32, 0, stream>>>(Qb, Kb, Vt, hg, als, Ab);
  out_gemm_kernel<<<dim3(72, 16), 256, 0, stream>>>(Ab, WpT, bp, Z);
  finalize_kernel<<<dim3((4 * 4 * 12 * 12 * DIMC + 255) / 256), 256, 0, stream>>>(
      Z, skip, (float*)d_out);

  (void)in_sizes; (void)n_in; (void)out_size; (void)ws_size;
}
